// DIFF_RWKV_86474871538071
// MI455X (gfx1250) — matrix operand order repaired and checked by reading, not compiled, not run on hardware
//
#include <hip/hip_runtime.h>
#include <hip/hip_bf16.h>

#define L_ 8
#define B_ 8
#define S_ 1024
#define D_ 768
#define H_ 3072
#define TC_ 128
#define NEG_ -1e30f

typedef unsigned short u16;
typedef __attribute__((ext_vector_type(16))) __bf16 v16bf;
typedef __attribute__((ext_vector_type(8)))  float  v8f;

__device__ __forceinline__ u16 f2bf(float f) {
  unsigned int u = __builtin_bit_cast(unsigned int, f);
  u += 0x7FFFu + ((u >> 16) & 1u);   // round-to-nearest-even
  return (u16)(u >> 16);
}

// ---------------------------------------------------------------- cast f32->bf16
__global__ void cast_bf16_k(const float* __restrict__ s, u16* __restrict__ d, long long n) {
  long long i = (long long)blockIdx.x * 256 + threadIdx.x;
  long long stride = (long long)gridDim.x * 256;
  for (; i < n; i += stride) d[i] = f2bf(s[i]);
}

// ---------------------------------------------------------------- time-embedding MLP: te[l][b][D]
__global__ __launch_bounds__(256) void te_k(const float* __restrict__ ts,
                                            const float* __restrict__ W1, const float* __restrict__ b1,
                                            const float* __restrict__ W2, const float* __restrict__ b2,
                                            float* __restrict__ te) {
  const int l = blockIdx.x / B_;
  const int b = blockIdx.x % B_;
  const int tid = threadIdx.x;
  __shared__ float temb[TC_];
  __shared__ float hid[4 * TC_];
  const float t = ts[b];
  for (int c = tid; c < TC_; c += 256) {
    const int half = TC_ / 2;                       // 64
    const float fac = -0.14619588f;                 // -log(10000)/63
    float fr = __expf((float)(c % half) * fac);
    float a = t * fr;
    temb[c] = (c < half) ? __cosf(a) : __sinf(a);
  }
  __syncthreads();
  const float* w1 = W1 + (size_t)l * TC_ * 4 * TC_;
  for (int j = tid; j < 4 * TC_; j += 256) {
    float s = b1[l * 4 * TC_ + j];
    for (int c = 0; c < TC_; c++) s += temb[c] * w1[c * 4 * TC_ + j];
    hid[j] = s / (1.f + __expf(-s));                // silu
  }
  __syncthreads();
  const float* w2 = W2 + (size_t)l * 4 * TC_ * D_;
  for (int d = tid; d < D_; d += 256) {
    float s = b2[l * D_ + d];
    for (int j = 0; j < 4 * TC_; j++) s += hid[j] * w2[j * D_ + d];
    te[((size_t)l * B_ + b) * D_ + d] = s;
  }
}

// ---------------------------------------------------------------- layernorm over D (optional +te[b])
__global__ __launch_bounds__(256) void ln_k(const float* __restrict__ x, const float* __restrict__ teL,
                                            const float* __restrict__ g, const float* __restrict__ bt,
                                            float* __restrict__ y) {
  __shared__ float red[256];
  const int tok = blockIdx.x;                       // 0..B*S-1
  const int b = tok / S_;
  const int tid = threadIdx.x;
  const float* xr = x + (size_t)tok * D_;
  float v[3];
#pragma unroll
  for (int k = 0; k < 3; k++) {
    int d = tid + k * 256;
    v[k] = xr[d] + (teL ? teL[b * D_ + d] : 0.f);
  }
  red[tid] = v[0] + v[1] + v[2];
  __syncthreads();
  for (int o = 128; o > 0; o >>= 1) { if (tid < o) red[tid] += red[tid + o]; __syncthreads(); }
  const float mean = red[0] * (1.f / D_);
  __syncthreads();
  float d0 = v[0] - mean, d1 = v[1] - mean, d2 = v[2] - mean;
  red[tid] = d0 * d0 + d1 * d1 + d2 * d2;
  __syncthreads();
  for (int o = 128; o > 0; o >>= 1) { if (tid < o) red[tid] += red[tid + o]; __syncthreads(); }
  const float rs = rsqrtf(red[0] * (1.f / D_) + 1e-5f);
  float* yr = y + (size_t)tok * D_;
#pragma unroll
  for (int k = 0; k < 3; k++) {
    int d = tid + k * 256;
    yr[d] = (v[k] - mean) * rs * g[d] + bt[d];
  }
}

// ---------------------------------------------------------------- token-shift + mix -> bf16 GEMM inputs
__global__ void mix_k(const float* __restrict__ xln, const float* __restrict__ shiftB,
                      const float* __restrict__ mk, const float* __restrict__ mv,
                      const float* __restrict__ mr,
                      u16* __restrict__ ok, u16* __restrict__ ov, u16* __restrict__ orr) {
  const long long total = (long long)B_ * S_ * D_;
  long long idx = (long long)blockIdx.x * 256 + threadIdx.x;
  const long long stride = (long long)gridDim.x * 256;
  for (; idx < total; idx += stride) {
    int d = (int)(idx % D_);
    long long token = idx / D_;
    int sPos = (int)(token % S_);
    int b = (int)(token / S_);
    float x = xln[idx];
    float xx = (sPos == 0) ? (shiftB ? shiftB[b * D_ + d] : 0.f) : xln[idx - D_];
    float m = mk[d];
    ok[idx] = f2bf(x * m + xx * (1.f - m));
    if (ov) { m = mv[d]; ov[idx] = f2bf(x * m + xx * (1.f - m)); }
    m = mr[d];
    orr[idx] = f2bf(x * m + xx * (1.f - m));
  }
}

// ---------------------------------------------------------------- RWKV-4 stabilized WKV scan (serial in S)
__global__ __launch_bounds__(256) void wkv_k(const float* __restrict__ k, const float* __restrict__ v,
                                             const float* __restrict__ first, const float* __restrict__ decay,
                                             const float* __restrict__ aa0, const float* __restrict__ bb0,
                                             const float* __restrict__ pp0, float* __restrict__ out) {
  const int gid = blockIdx.x * 256 + threadIdx.x;   // < B*D
  const int d = gid % D_;
  const int b = gid / D_;
  const float u = first[d];
  const float w = -__expf(decay[d]);
  float aa = aa0 ? aa0[gid] : 0.f;
  float bb = bb0 ? bb0[gid] : 0.f;
  float pp = pp0 ? pp0[gid] : NEG_;
  size_t idx = (size_t)b * S_ * D_ + d;
  for (int s = 0; s < S_; s++, idx += D_) {
    float kt = k[idx], vt = v[idx];
    if (s + 8 < S_) {                               // global_prefetch_b8 hints down the stride
      __builtin_prefetch(&k[idx + 8 * D_], 0, 0);
      __builtin_prefetch(&v[idx + 8 * D_], 0, 0);
    }
    float ww = u + kt;
    float p = fmaxf(pp, ww);
    float e1 = __expf(pp - p), e2 = __expf(ww - p);
    out[idx] = (e1 * aa + e2 * vt) / (e1 * bb + e2);
    float ww2 = pp + w;
    float p2 = fmaxf(ww2, kt);
    e1 = __expf(ww2 - p2); e2 = __expf(kt - p2);
    aa = e1 * aa + e2 * vt; bb = e1 * bb + e2; pp = p2;
  }
}

// ---------------------------------------------------------------- r * wkv -> bf16
__global__ void rmul_k(const float* __restrict__ r, const float* __restrict__ wkv,
                       u16* __restrict__ out) {
  const long long total = (long long)B_ * S_ * D_;
  long long idx = (long long)blockIdx.x * 256 + threadIdx.x;
  const long long stride = (long long)gridDim.x * 256;
  for (; idx < total; idx += stride) out[idx] = f2bf(r[idx] * wkv[idx]);
}

// ---------------------------------------------------------------- bf16 WMMA GEMM, C = A[M,K] @ W[K,N]
// 128x128 block tile, K-step 32, double-buffered LDS.
// A tile staged with GLOBAL_LOAD_ASYNC_TO_LDS_B128 (ASYNCcnt path, no VGPR bounce);
// B tile staged through registers with K-pair interleave so each fragment VGPR is one dword.
// 8 waves (wave32): wave = 64(M) x 32(N) = 4x2 fragments.
// MODE: 0 = f32 store, 1 = sigmoid f32, 2 = relu^2 -> bf16, 3 = Cf += acc, 4 = Cf += R*acc
template <int MODE>
__global__ __launch_bounds__(256) void gemm_k(const u16* __restrict__ A, const u16* __restrict__ W,
                                              int N, int K,
                                              float* __restrict__ Cf, u16* __restrict__ Cb,
                                              const float* __restrict__ R) {
  __shared__ __align__(16) u16 lA[2][128 * 32];              // A tiles, row-major [128][32]
  __shared__ __align__(16) unsigned int lB[2][16 * 128];     // B tiles, K-pair-interleaved [16][128]
  const int tid  = threadIdx.x;
  const int lane = tid & 31;
  const int wave = tid >> 5;
  const int wm = wave >> 2;          // 0..1  (M half of block tile)
  const int wn = wave & 3;           // 0..3  (N quarter)
  const int hi = lane >> 4;          // lane half per ISA fragment tables
  const int lq = lane & 15;
  const int m0 = blockIdx.y * 128;
  const int n0 = blockIdx.x * 128;

  v8f acc[4][2] = {};

  const int pm = tid >> 4;           // B pair-row 0..15
  const int pc = (tid & 15) * 8;     // B col group (8 cols)

  // ---- staging helpers ------------------------------------------------------
  auto stageA_async = [&](int nb, int ktn) {
#pragma unroll
    for (int it = 0; it < 2; it++) {
      int f = tid + it * 256;                    // 16B chunk 0..511
      int row = f >> 2;
      int col = (f & 3) * 8;
      unsigned ldsOff = (unsigned)(unsigned long long)(const void*)&lA[nb][f * 8];
      const u16* g = A + (size_t)(m0 + row) * K + ktn + col;
      // per-lane 16B global -> LDS, tracked by ASYNCcnt
      asm volatile("global_load_async_to_lds_b128 %0, %1, off"
                   :: "v"(ldsOff), "v"(g) : "memory");
    }
  };
  auto loadB = [&](int ktn, uint4& ra, uint4& rb) {
    const u16* w0 = W + (size_t)(ktn + 2 * pm) * N + n0 + pc;
    ra = *(const uint4*)w0;                      // row 2m,   cols pc..pc+7
    rb = *(const uint4*)(w0 + N);                // row 2m+1, cols pc..pc+7
  };
  auto storeB = [&](int nb, uint4 a, uint4 bq) {
    unsigned int o[8];
    o[0] = (a.x & 0xFFFFu) | (bq.x << 16);
    o[1] = (a.x >> 16)     | (bq.x & 0xFFFF0000u);
    o[2] = (a.y & 0xFFFFu) | (bq.y << 16);
    o[3] = (a.y >> 16)     | (bq.y & 0xFFFF0000u);
    o[4] = (a.z & 0xFFFFu) | (bq.z << 16);
    o[5] = (a.z >> 16)     | (bq.z & 0xFFFF0000u);
    o[6] = (a.w & 0xFFFFu) | (bq.w << 16);
    o[7] = (a.w >> 16)     | (bq.w & 0xFFFF0000u);
    unsigned int* dst = &lB[nb][pm * 128 + pc];
    ((uint4*)dst)[0] = make_uint4(o[0], o[1], o[2], o[3]);
    ((uint4*)dst)[1] = make_uint4(o[4], o[5], o[6], o[7]);
  };

  // ---- prologue: stage buffer 0 --------------------------------------------
  {
    stageA_async(0, 0);
    uint4 ra, rb;
    loadB(0, ra, rb);
    storeB(0, ra, rb);
    asm volatile("s_wait_asynccnt 0x0" ::: "memory");
    __syncthreads();
  }

  for (int kt = 0; kt < K; kt += 32) {
    const int buf = (kt >> 5) & 1;
    const int nbuf = buf ^ 1;
    const bool hasNext = (kt + 32) < K;
    uint4 ra, rb;
    if (hasNext) {
      stageA_async(nbuf, kt + 32);               // async, overlaps with WMMA below
      loadB(kt + 32, ra, rb);                    // global loads in flight over compute
    }

    // ---- fragment loads for A 16x32 / B 32x16, contiguous K order k = 16*hi + s in both A and B
    const uint4* lAq = (const uint4*)lA[buf];
    const unsigned int* lBb = lB[buf];
    union FragA { uint4 q[2]; v16bf v; } fa[4];
    union FragB { unsigned int d[8]; v16bf v; } fb[2];
#pragma unroll
    for (int fm = 0; fm < 4; fm++) {
      int row = wm * 64 + fm * 16 + lq;                 // A row = M
      fa[fm].q[0] = lAq[row * 4 + 2 * hi];              // K = {hi*16 .. hi*16+7}
      fa[fm].q[1] = lAq[row * 4 + 2 * hi + 1];          // K = {hi*16+8 .. hi*16+15}
    }
#pragma unroll
    for (int fn = 0; fn < 2; fn++) {
      int n = wn * 32 + fn * 16 + lq;
#pragma unroll
      for (int j = 0; j < 8; j++)
        fb[fn].d[j] = lBb[(hi * 8 + j) * 128 + n];      // K pair {16*hi+2j, +1}, col n
    }
#pragma unroll
    for (int fm = 0; fm < 4; fm++)
#pragma unroll
      for (int fn = 0; fn < 2; fn++)
        acc[fm][fn] = __builtin_amdgcn_wmma_f32_16x16x32_bf16(
            false, fa[fm].v, false, fb[fn].v, (short)0, acc[fm][fn], false, false);

    if (hasNext) storeB(nbuf, ra, rb);                  // ds_store after WMMAs
    asm volatile("s_wait_asynccnt 0x0" ::: "memory");   // async LDS writes visible
    __syncthreads();
  }

  // ---- epilogue (C 16x16 f32 layout: VGPR r -> M = hi*8 + r, N = lq)
#pragma unroll
  for (int fm = 0; fm < 4; fm++) {
#pragma unroll
    for (int fn = 0; fn < 2; fn++) {
      int gn = n0 + wn * 32 + fn * 16 + lq;
#pragma unroll
      for (int r = 0; r < 8; r++) {
        int gm = m0 + wm * 64 + fm * 16 + hi * 8 + r;
        size_t o = (size_t)gm * N + gn;
        float val = acc[fm][fn][r];
        if (MODE == 0) Cf[o] = val;
        else if (MODE == 1) Cf[o] = 1.f / (1.f + __expf(-val));
        else if (MODE == 2) { float t = fmaxf(val, 0.f); Cb[o] = f2bf(t * t); }
        else if (MODE == 3) Cf[o] += val;
        else if (MODE == 4) Cf[o] += R[o] * val;
      }
    }
  }
}

// ================================================================ host orchestration
extern "C" void kernel_launch(void* const* d_in, const int* in_sizes, int n_in,
                              void* d_out, int out_size, void* d_ws, size_t ws_size,
                              hipStream_t stream) {
  (void)in_sizes; (void)n_in; (void)out_size; (void)ws_size;
  const float* x_t       = (const float*)d_in[0];
  const float* timesteps = (const float*)d_in[1];
  const float* enc_shift = (const float*)d_in[2];
  const float* enc_aa    = (const float*)d_in[3];
  const float* enc_bb    = (const float*)d_in[4];
  const float* enc_pp    = (const float*)d_in[5];
  const float* ln0_g     = (const float*)d_in[6];
  const float* ln0_b     = (const float*)d_in[7];
  const float* ln_g      = (const float*)d_in[8];
  const float* ln_b      = (const float*)d_in[9];
  const float* lnout_g   = (const float*)d_in[10];
  const float* lnout_b   = (const float*)d_in[11];
  const float* att_decay = (const float*)d_in[12];
  const float* att_first = (const float*)d_in[13];
  const float* att_mix   = (const float*)d_in[14];
  const float* att_W     = (const float*)d_in[15];
  const float* ffn_mix   = (const float*)d_in[16];
  const float* ffn_Wk    = (const float*)d_in[17];
  const float* ffn_Wv    = (const float*)d_in[18];
  const float* ffn_Wr    = (const float*)d_in[19];
  const float* tp_W1     = (const float*)d_in[20];
  const float* tp_b1     = (const float*)d_in[21];
  const float* tp_W2     = (const float*)d_in[22];
  const float* tp_b2     = (const float*)d_in[23];

  char* ws = (char*)d_ws;
  size_t off = 0;
  auto alloc = [&](size_t bytes) -> void* {
    void* p = ws + off;
    off += (bytes + 255) & ~(size_t)255;
    return p;
  };
  const size_t TOK = (size_t)B_ * S_;            // 8192
  const size_t TOT = TOK * D_;                   // 6.29M
  u16*  attW_bf = (u16*)alloc((size_t)L_ * 2 * 4 * D_ * D_ * 2);
  u16*  Wk_bf   = (u16*)alloc((size_t)L_ * D_ * H_ * 2);
  u16*  Wv_bf   = (u16*)alloc((size_t)L_ * H_ * D_ * 2);
  u16*  Wr_bf   = (u16*)alloc((size_t)L_ * D_ * D_ * 2);
  float* h      = (float*)alloc(TOT * 4);
  float* xln    = (float*)alloc(TOT * 4);
  float* te     = (float*)alloc((size_t)L_ * B_ * D_ * 4);
  u16*  xmk     = (u16*)alloc(TOT * 2);
  u16*  xmv     = (u16*)alloc(TOT * 2);
  u16*  xmr     = (u16*)alloc(TOT * 2);
  float* kbuf   = (float*)alloc(TOT * 4);
  float* vbuf   = (float*)alloc(TOT * 4);
  float* rbuf   = (float*)alloc(TOT * 4);
  float* wkvbuf = (float*)alloc(TOT * 4);
  u16*  rwkv_bf = (u16*)alloc(TOT * 2);
  u16*  kk_bf   = (u16*)alloc(TOK * H_ * 2);

  // weight casts (graph-captured; deterministic)
  cast_bf16_k<<<4096, 256, 0, stream>>>(att_W,  attW_bf, (long long)L_ * 2 * 4 * D_ * D_);
  cast_bf16_k<<<4096, 256, 0, stream>>>(ffn_Wk, Wk_bf,   (long long)L_ * D_ * H_);
  cast_bf16_k<<<4096, 256, 0, stream>>>(ffn_Wv, Wv_bf,   (long long)L_ * H_ * D_);
  cast_bf16_k<<<2048, 256, 0, stream>>>(ffn_Wr, Wr_bf,   (long long)L_ * D_ * D_);
  te_k<<<L_ * B_, 256, 0, stream>>>(timesteps, tp_W1, tp_b1, tp_W2, tp_b2, te);
  // layer-0 pre-LN: h = ln0(x_t)
  ln_k<<<(int)TOK, 256, 0, stream>>>(x_t, nullptr, ln0_g, ln0_b, h);

  auto gemm = [&](int mode, const u16* A, const u16* W, int N, int K,
                  float* Cf, u16* Cb, const float* R) {
    dim3 grid(N / 128, (int)(TOK / 128));
    switch (mode) {
      case 0: gemm_k<0><<<grid, 256, 0, stream>>>(A, W, N, K, Cf, Cb, R); break;
      case 1: gemm_k<1><<<grid, 256, 0, stream>>>(A, W, N, K, Cf, Cb, R); break;
      case 2: gemm_k<2><<<grid, 256, 0, stream>>>(A, W, N, K, Cf, Cb, R); break;
      case 3: gemm_k<3><<<grid, 256, 0, stream>>>(A, W, N, K, Cf, Cb, R); break;
      case 4: gemm_k<4><<<grid, 256, 0, stream>>>(A, W, N, K, Cf, Cb, R); break;
    }
  };

  const int mixGrid = (int)((TOT + 255) / 256);

  for (int i = 0; i < L_; i++) {
    // ---------- time-mix block, channel 0 (zero initial state) ----------
    ln_k<<<(int)TOK, 256, 0, stream>>>(h, te + (size_t)i * B_ * D_,
                                       ln_g + (size_t)(i * 3 + 0) * D_, ln_b + (size_t)(i * 3 + 0) * D_, xln);
    {
      const float* mx = att_mix + (size_t)((i * 2 + 0) * 3) * D_;
      mix_k<<<mixGrid, 256, 0, stream>>>(xln, nullptr, mx, mx + D_, mx + 2 * D_, xmk, xmv, xmr);
      const u16* W = attW_bf + (size_t)((i * 2 + 0) * 4) * D_ * D_;
      gemm(0, xmk, W + 0 * (size_t)D_ * D_, D_, D_, kbuf, nullptr, nullptr);
      gemm(0, xmv, W + 1 * (size_t)D_ * D_, D_, D_, vbuf, nullptr, nullptr);
      gemm(1, xmr, W + 2 * (size_t)D_ * D_, D_, D_, rbuf, nullptr, nullptr);
      wkv_k<<<(B_ * D_) / 256, 256, 0, stream>>>(kbuf, vbuf,
          att_first + (size_t)(i * 2 + 0) * D_, att_decay + (size_t)(i * 2 + 0) * D_,
          nullptr, nullptr, nullptr, wkvbuf);
      rmul_k<<<mixGrid, 256, 0, stream>>>(rbuf, wkvbuf, rwkv_bf);
      gemm(3, rwkv_bf, W + 3 * (size_t)D_ * D_, D_, D_, h, nullptr, nullptr);   // h += att
    }
    // ---------- time-mix block, channel 1 (encoder state) ----------
    ln_k<<<(int)TOK, 256, 0, stream>>>(h, nullptr,
                                       ln_g + (size_t)(i * 3 + 1) * D_, ln_b + (size_t)(i * 3 + 1) * D_, xln);
    {
      const float* mx = att_mix + (size_t)((i * 2 + 1) * 3) * D_;
      mix_k<<<mixGrid, 256, 0, stream>>>(xln, enc_shift + (size_t)i * B_ * D_,
                                         mx, mx + D_, mx + 2 * D_, xmk, xmv, xmr);
      const u16* W = attW_bf + (size_t)((i * 2 + 1) * 4) * D_ * D_;
      gemm(0, xmk, W + 0 * (size_t)D_ * D_, D_, D_, kbuf, nullptr, nullptr);
      gemm(0, xmv, W + 1 * (size_t)D_ * D_, D_, D_, vbuf, nullptr, nullptr);
      gemm(1, xmr, W + 2 * (size_t)D_ * D_, D_, D_, rbuf, nullptr, nullptr);
      wkv_k<<<(B_ * D_) / 256, 256, 0, stream>>>(kbuf, vbuf,
          att_first + (size_t)(i * 2 + 1) * D_, att_decay + (size_t)(i * 2 + 1) * D_,
          enc_aa + (size_t)i * B_ * D_, enc_bb + (size_t)i * B_ * D_, enc_pp + (size_t)i * B_ * D_,
          wkvbuf);
      rmul_k<<<mixGrid, 256, 0, stream>>>(rbuf, wkvbuf, rwkv_bf);
      gemm(3, rwkv_bf, W + 3 * (size_t)D_ * D_, D_, D_, h, nullptr, nullptr);   // h += att
    }
    // ---------- channel-mix ----------
    ln_k<<<(int)TOK, 256, 0, stream>>>(h, nullptr,
                                       ln_g + (size_t)(i * 3 + 2) * D_, ln_b + (size_t)(i * 3 + 2) * D_, xln);
    {
      const float* fm = ffn_mix + (size_t)(i * 2) * D_;
      mix_k<<<mixGrid, 256, 0, stream>>>(xln, nullptr, fm, nullptr, fm + D_, xmk, nullptr, xmr);
      gemm(2, xmk, Wk_bf + (size_t)i * D_ * H_, H_, D_, nullptr, kk_bf, nullptr);  // relu^2 -> bf16
      gemm(1, xmr, Wr_bf + (size_t)i * D_ * D_, D_, D_, rbuf, nullptr, nullptr);   // sigmoid
      gemm(4, kk_bf, Wv_bf + (size_t)i * H_ * D_, D_, H_, h, nullptr, rbuf);       // h += r * (k @ Wv)
    }
  }

  ln_k<<<(int)TOK, 256, 0, stream>>>(h, nullptr, lnout_g, lnout_b, (float*)d_out);
}
